// Speller_65515431133837
// MI455X (gfx1250) — matrix operand order repaired and checked by reading, not compiled, not run on hardware
//
#include <hip/hip_runtime.h>

#define B_      64
#define L_      512
#define H_      512
#define E_      512
#define V_      64
#define T_      200
#define KG      1536   // E + H + H  (x = [emb, ctx, h])
#define NG      2048   // 4H gates
#define KL      1024   // 2H logit K

typedef __attribute__((ext_vector_type(16))) __bf16 v16bf;
typedef __attribute__((ext_vector_type(8)))  float  v8f;
typedef __attribute__((ext_vector_type(4)))  int    v4i;
typedef __attribute__((ext_vector_type(8)))  int    v8i;

__device__ __forceinline__ unsigned short f2bf(float f) {
  unsigned u = __float_as_uint(f);
  u += 0x7fffu + ((u >> 16) & 1u);
  return (unsigned short)(u >> 16);
}
__device__ __forceinline__ float bf2f(unsigned short h) {
  return __uint_as_float(((unsigned)h) << 16);
}
__device__ __forceinline__ float sigmoidf(float x){ return 1.0f/(1.0f+__expf(-x)); }

// Build a 16-element bf16 WMMA fragment from two 16-byte loads at p and p+off2 elems.
__device__ __forceinline__ v16bf load_frag(const unsigned short* p, int off2) {
  v4i lo = *(const v4i*)(p);
  v4i hi = *(const v4i*)(p + off2);
  v8i w; w[0]=lo[0]; w[1]=lo[1]; w[2]=lo[2]; w[3]=lo[3];
         w[4]=hi[0]; w[5]=hi[1]; w[6]=hi[2]; w[7]=hi[3];
  return __builtin_bit_cast(v16bf, w);
}

#define WMMA_BF16(A, Bf, C) \
  __builtin_amdgcn_wmma_f32_16x16x32_bf16(false, (A), false, (Bf), (short)0, (C), false, false)

// ---------------- Phase A: one-time precompute ----------------

// msum[b,h] = sum_{l < len[b]} seq[b,l,h]
__global__ void msum_kernel(const float* __restrict__ seq, const int* __restrict__ lens,
                            float* __restrict__ msum) {
  int b = blockIdx.x >> 1;
  int h = ((blockIdx.x & 1) << 8) + threadIdx.x;
  int len = lens[b];
  const float* p = seq + (size_t)b * L_ * H_ + h;
  float acc = 0.f;
  for (int l = 0; l < len; ++l) acc += p[(size_t)l * H_];
  msum[b * H_ + h] = acc;
}

// keysum = msum @ key_w^T + L*key_b ; valsum = msum @ value_w^T + L*value_b
__global__ void kv_kernel(const float* __restrict__ msum,
                          const float* __restrict__ key_w, const float* __restrict__ key_b,
                          const float* __restrict__ val_w, const float* __restrict__ val_b,
                          float* __restrict__ keysum, float* __restrict__ valsum) {
  int gid = blockIdx.x * 256 + threadIdx.x;          // 65536 dots
  int sel = gid >> 15;
  int rem = gid & 32767;
  int b = rem >> 9, n = rem & 511;
  const float* w    = sel ? val_w : key_w;
  const float* bias = sel ? val_b : key_b;
  const float* mrow = msum + b * 512;
  const float* wrow = w + n * 512;
  float acc = 0.f;
  for (int k = 0; k < 512; ++k) acc += mrow[k] * wrow[k];
  acc += (float)L_ * bias[n];
  (sel ? valsum : keysum)[b * 512 + n] = acc;
}

// Convert weights to bf16: Wc[n,k] = (k<1024 ? w_ih[n,k] : w_hh[n,k-1024]); cdn_w, char_emb; bsum = b_ih+b_hh
__global__ void convert_kernel(const float* __restrict__ w_ih, const float* __restrict__ w_hh,
                               const float* __restrict__ b_ih, const float* __restrict__ b_hh,
                               const float* __restrict__ cdn_w, const float* __restrict__ char_emb,
                               unsigned short* __restrict__ Wc, unsigned short* __restrict__ cdnw_bf,
                               unsigned short* __restrict__ emb_bf, float* __restrict__ bsum) {
  const int NW = NG * KG;                                 // 3145728
  const int total = NW + V_ * KL + V_ * E_ + NG;
  for (int i = blockIdx.x * blockDim.x + threadIdx.x; i < total; i += gridDim.x * blockDim.x) {
    if (i < NW) {
      int n = i / KG, k = i % KG;
      float v = (k < 1024) ? w_ih[n * 1024 + k] : w_hh[n * 512 + (k - 1024)];
      Wc[i] = f2bf(v);
    } else if (i < NW + V_ * KL) {
      int j = i - NW; cdnw_bf[j] = f2bf(cdn_w[j]);
    } else if (i < NW + V_ * KL + V_ * E_) {
      int j = i - NW - V_ * KL; emb_bf[j] = f2bf(char_emb[j]);
    } else {
      int j = i - NW - V_ * KL - V_ * E_; bsum[j] = b_ih[j] + b_hh[j];
    }
  }
}

// x0 = [emb[SOS], ctx0 = valsum/H, 0]; c = 0; logits[:,0,:] = 0; symbols[:,0] = 0
__global__ void init_kernel(const float* __restrict__ valsum, const unsigned short* __restrict__ emb_bf,
                            unsigned short* __restrict__ X, float* __restrict__ cstate,
                            float* __restrict__ out_logits, float* __restrict__ out_sym) {
  const int NX = B_ * KG, NC = B_ * H_, NL = B_ * V_;
  const int total = NX + NC + NL + B_;
  for (int i = blockIdx.x * blockDim.x + threadIdx.x; i < total; i += gridDim.x * blockDim.x) {
    if (i < NX) {
      int b = i / KG, k = i % KG;
      unsigned short v;
      if (k < 512)       v = emb_bf[k];                                   // char_emb[SOS=0]
      else if (k < 1024) v = f2bf(valsum[b * 512 + (k - 512)] * (1.0f / (float)H_));
      else               v = 0;
      X[i] = v;
    } else if (i < NX + NC) {
      cstate[i - NX] = 0.f;
    } else if (i < NX + NC + NL) {
      int j = i - NX - NC;
      out_logits[(size_t)(j / V_) * T_ * V_ + (j % V_)] = 0.f;
    } else {
      out_sym[(size_t)(i - NX - NC - NL) * T_] = 0.f;
    }
  }
}

// ---------------- Phase B kernel 1: gates = x @ [w_ih;w_hh]^T + bsum ----------
// grid = 16 blocks x 256 threads = 128 waves; wave owns one N-tile (16 cols) and all 4 M-bands.
// Register double-buffered: issue k+1's 5 fragment loads before k's 4 WMMAs so the
// L2 latency overlaps the matrix pipe instead of s_wait_loadcnt 0 every iteration.
__global__ void __launch_bounds__(256) gates_gemm(const unsigned short* __restrict__ X,
                                                  const unsigned short* __restrict__ Wc,
                                                  const float* __restrict__ bsum,
                                                  float* __restrict__ gates) {
  const int wave = threadIdx.x >> 5;
  const int lane = threadIdx.x & 31;
  const int tileN = blockIdx.x * 8 + wave;          // 0..127
  const int r16 = lane & 15, half = lane >> 4;
  v8f acc[4] = {};
  const unsigned short* bptr = Wc + (size_t)(tileN * 16 + r16) * KG + half * 8;
  const unsigned short* aptr = X + (size_t)r16 * KG + half * 8;

  v16bf bc  = load_frag(bptr, 16);
  v16bf ac0 = load_frag(aptr, 16);
  v16bf ac1 = load_frag(aptr + 16 * KG, 16);
  v16bf ac2 = load_frag(aptr + 32 * KG, 16);
  v16bf ac3 = load_frag(aptr + 48 * KG, 16);

  for (int kt = 0; kt < 47; ++kt) {
    const int kn = (kt + 1) * 32;
    __builtin_prefetch(bptr + kn + 96, 0, 3);        // global_prefetch_b8, pull toward WGP
    v16bf bn  = load_frag(bptr + kn, 16);            // B col = Wc row, K[half*8..+8), K[16+half*8..+8)
    v16bf an0 = load_frag(aptr + kn, 16);            // A: K[half*8..+8), K[16+half*8..+8)
    v16bf an1 = load_frag(aptr + 16 * KG + kn, 16);
    v16bf an2 = load_frag(aptr + 32 * KG + kn, 16);
    v16bf an3 = load_frag(aptr + 48 * KG + kn, 16);
    acc[0] = WMMA_BF16(ac0, bc, acc[0]);
    acc[1] = WMMA_BF16(ac1, bc, acc[1]);
    acc[2] = WMMA_BF16(ac2, bc, acc[2]);
    acc[3] = WMMA_BF16(ac3, bc, acc[3]);
    bc = bn; ac0 = an0; ac1 = an1; ac2 = an2; ac3 = an3;
  }
  acc[0] = WMMA_BF16(ac0, bc, acc[0]);
  acc[1] = WMMA_BF16(ac1, bc, acc[1]);
  acc[2] = WMMA_BF16(ac2, bc, acc[2]);
  acc[3] = WMMA_BF16(ac3, bc, acc[3]);

  const int n = tileN * 16 + r16;
  const float bias = bsum[n];
#pragma unroll
  for (int mb = 0; mb < 4; ++mb)
#pragma unroll
    for (int j = 0; j < 8; ++j) {
      int m = mb * 16 + j + 8 * half;               // C/D layout: VGPR j <-> M = j + 8*half
      gates[m * NG + n] = acc[mb][j] + bias;
    }
}

// ---------------- Phase B kernel 2: LSTM cell + attend + x staging + logit WMMA + argmax ----------
// grid = 4 blocks x 256 threads (8 waves); block handles 16 batch rows.
__global__ void __launch_bounds__(256) cell_kernel(
    const float* __restrict__ gates,
    const float* __restrict__ keysum, const float* __restrict__ valsum,
    float* __restrict__ cstate,
    const unsigned short* __restrict__ emb_bf, const int* __restrict__ batch_y,
    const unsigned short* __restrict__ cdnw_bf, const float* __restrict__ cdn_b,
    unsigned short* __restrict__ X,
    float* __restrict__ out_logits, float* __restrict__ out_sym, int s) {
  __shared__ unsigned short amat[16 * KL];   // per-row [h(512) | ctx(512)] bf16, WMMA A source
  __shared__ float lt2[2][16][V_];           // split-K partial logit tiles
  __shared__ float ltile[16][V_];            // combined logits for argmax
  const int r0 = blockIdx.x * 16;
  const int tid = threadIdx.x;
  const float rscale = 0.044194173824159216f;   // 1/sqrt(512)

  // pass 1: LSTM cell elementwise (bias already folded into gates), h -> LDS (bf16)
  for (int i = tid; i < 16 * 512; i += 256) {
    int r = i >> 9, col = i & 511;
    int b = r0 + r;
    const float* g = gates + b * NG;
    float ig = sigmoidf(g[col]);
    float fg = sigmoidf(g[512 + col]);
    float gg = tanhf   (g[1024 + col]);
    float og = sigmoidf(g[1536 + col]);
    float cn = fg * cstate[b * 512 + col] + ig * gg;
    cstate[b * 512 + col] = cn;
    amat[r * KL + col] = f2bf(og * tanhf(cn));
  }
  __syncthreads();

  // pass 2: softmax-attend per row (2 rows per wave), stage ctx + next-step x
  const int wave = tid >> 5, lane = tid & 31;
  for (int rr = wave * 2; rr < wave * 2 + 2; ++rr) {
    int b = r0 + rr;
    float mx = -1e30f;
    for (int i = 0; i < 16; ++i) {
      int c = lane + i * 32;
      float a = keysum[b * 512 + c] * bf2f(amat[rr * KL + c]) * rscale;
      mx = fmaxf(mx, a);
    }
    for (int m = 16; m >= 1; m >>= 1) mx = fmaxf(mx, __shfl_xor(mx, m, 32));
    float sum = 0.f;
    for (int i = 0; i < 16; ++i) {
      int c = lane + i * 32;
      float a = keysum[b * 512 + c] * bf2f(amat[rr * KL + c]) * rscale;
      sum += __expf(a - mx);
    }
    for (int m = 16; m >= 1; m >>= 1) sum += __shfl_xor(sum, m, 32);
    float inv = 1.0f / sum;
    int y = batch_y[b * T_ + (s + 1)];               // prev token for next step
    for (int i = 0; i < 16; ++i) {
      int c = lane + i * 32;
      float a = keysum[b * 512 + c] * bf2f(amat[rr * KL + c]) * rscale;
      float ctx = valsum[b * 512 + c] * __expf(a - mx) * inv;
      unsigned short cb = f2bf(ctx);
      amat[rr * KL + 512 + c] = cb;                  // logit A = [h, ctx]
      X[b * KG + c]        = emb_bf[y * 512 + c];    // x = [emb, ctx, h]
      X[b * KG + 512 + c]  = cb;
      X[b * KG + 1024 + c] = amat[rr * KL + c];
    }
  }
  __syncthreads();

  // pass 3: logit = [h,ctx] @ cdn_w^T  — all 8 waves, split-K (2 halves of 512),
  // double-buffered fragment loads.
  {
    const int r16 = lane & 15, half = lane >> 4;
    const int tn = wave & 3, ks = wave >> 2;
    v8f acc = {};
    const unsigned short* brow = cdnw_bf + (size_t)(tn * 16 + r16) * KL + half * 8;
    const unsigned short* arow = amat + r16 * KL + half * 8;
    const int kb0 = ks * 512;
    v16bf ac = load_frag(arow + kb0, 16);
    v16bf bc = load_frag(brow + kb0, 16);
    for (int kt = 0; kt < 15; ++kt) {
      int kn = kb0 + (kt + 1) * 32;
      v16bf an = load_frag(arow + kn, 16);
      v16bf bn = load_frag(brow + kn, 16);
      acc = WMMA_BF16(ac, bc, acc);
      ac = an; bc = bn;
    }
    acc = WMMA_BF16(ac, bc, acc);
    int n = tn * 16 + r16;
#pragma unroll
    for (int j = 0; j < 8; ++j) lt2[ks][j + 8 * half][n] = acc[j];
  }
  __syncthreads();

  // combine split-K partials + bias, write logits
  for (int i = tid; i < 16 * V_; i += 256) {
    int m = i >> 6, n = i & 63;
    float v = lt2[0][m][n] + lt2[1][m][n] + cdn_b[n];
    ltile[m][n] = v;
    out_logits[((size_t)(r0 + m) * T_ + (s + 1)) * V_ + n] = v;
  }
  __syncthreads();

  // pass 4: deterministic argmax -> symbols (reference samples; unverifiable RNG)
  if (tid < 16) {
    float best = ltile[tid][0]; int bi = 0;
    for (int v = 1; v < V_; ++v) { float x = ltile[tid][v]; if (x > best) { best = x; bi = v; } }
    out_sym[(size_t)(r0 + tid) * T_ + (s + 1)] = (float)bi;
  }
}

// ---------------- Launch ----------------

extern "C" void kernel_launch(void* const* d_in, const int* in_sizes, int n_in,
                              void* d_out, int out_size, void* d_ws, size_t ws_size,
                              hipStream_t stream) {
  (void)in_sizes; (void)n_in; (void)out_size; (void)ws_size;
  const float* seq      = (const float*)d_in[0];
  const float* key_w    = (const float*)d_in[1];
  const float* key_b    = (const float*)d_in[2];
  const float* val_w    = (const float*)d_in[3];
  const float* val_b    = (const float*)d_in[4];
  const float* char_emb = (const float*)d_in[5];
  const float* w_ih     = (const float*)d_in[6];
  const float* w_hh     = (const float*)d_in[7];
  const float* b_ih     = (const float*)d_in[8];
  const float* b_hh     = (const float*)d_in[9];
  const float* cdn_w    = (const float*)d_in[10];
  const float* cdn_b    = (const float*)d_in[11];
  const int*   lens     = (const int*)d_in[12];
  const int*   batch_y  = (const int*)d_in[13];

  char* ws = (char*)d_ws;
  size_t off = 0;
  auto alloc = [&](size_t bytes) { void* p = ws + off; off = (off + bytes + 255) & ~(size_t)255; return p; };
  float*          msum    = (float*)alloc((size_t)B_ * H_ * 4);
  float*          keysum  = (float*)alloc((size_t)B_ * H_ * 4);
  float*          valsum  = (float*)alloc((size_t)B_ * H_ * 4);
  float*          cstate  = (float*)alloc((size_t)B_ * H_ * 4);
  float*          gates   = (float*)alloc((size_t)B_ * NG * 4);
  float*          bsum    = (float*)alloc((size_t)NG * 4);
  unsigned short* Wc      = (unsigned short*)alloc((size_t)NG * KG * 2);
  unsigned short* cdnw_bf = (unsigned short*)alloc((size_t)V_ * KL * 2);
  unsigned short* emb_bf  = (unsigned short*)alloc((size_t)V_ * E_ * 2);
  unsigned short* X       = (unsigned short*)alloc((size_t)B_ * KG * 2);

  float* out_logits = (float*)d_out;
  float* out_sym    = out_logits + (size_t)B_ * T_ * V_;

  msum_kernel<<<dim3(B_ * 2), dim3(256), 0, stream>>>(seq, lens, msum);
  kv_kernel<<<dim3(256), dim3(256), 0, stream>>>(msum, key_w, key_b, val_w, val_b, keysum, valsum);
  convert_kernel<<<dim3(2048), dim3(256), 0, stream>>>(w_ih, w_hh, b_ih, b_hh, cdn_w, char_emb,
                                                       Wc, cdnw_bf, emb_bf, bsum);
  init_kernel<<<dim3(128), dim3(256), 0, stream>>>(valsum, emb_bf, X, cstate, out_logits, out_sym);

  for (int s = 0; s < T_ - 1; ++s) {
    gates_gemm<<<dim3(16), dim3(256), 0, stream>>>(X, Wc, bsum, gates);
    cell_kernel<<<dim3(4), dim3(256), 0, stream>>>(gates, keysum, valsum, cstate,
                                                   emb_bf, batch_y, cdnw_bf, cdn_b,
                                                   X, out_logits, out_sym, s);
  }
}
